// WeightedMultiHeadedAttention_14370960572821
// MI455X (gfx1250) — matrix operand order repaired and checked by reading, not compiled, not run on hardware
//
#include <hip/hip_runtime.h>
#include <hip/hip_bf16.h>

// MI455X (gfx1250) weighted multi-head attention.
// B=2, S=2048, D=512, H=8, DK=64.
// attn_weight (268 MB f32, read exactly once) dominates HBM traffic -> the
// kernel is bandwidth bound (~11.5us floor at 23.3 TB/s). All matmuls use
// bf16 WMMA with f32 accumulation (v_wmma_f32_16x16x32_bf16). Flash-style
// online softmax keeps the SxS scores out of HBM. K/V tiles are staged into
// double-buffered LDS by the Tensor Data Mover (tensor_load_to_lds +
// s_wait_tensorcnt), with TDM LDS-padding to bank-stagger the rows.

#define BB   2
#define SS   2048
#define DDm  512
#define HHn  8
#define DKK  64
#define MTOT (BB * SS) // 4096

typedef __attribute__((ext_vector_type(16))) __bf16 v16bf;
typedef __attribute__((ext_vector_type(8)))  __bf16 v8bf;
typedef __attribute__((ext_vector_type(8)))  float  v8f;
typedef __attribute__((ext_vector_type(4)))  unsigned int u32x4;
typedef __attribute__((ext_vector_type(8)))  int i32x8;
typedef __attribute__((ext_vector_type(4)))  int i32x4;

// Padded LDS row strides (elements) produced by TDM pad_enable:
// K tile rows: 64 bf16 = 128B + 16B pad -> 72 elements.
// V tile rows: 32 bf16 =  64B + 16B pad -> 40 elements.
#define KLD 72
#define VLD 40

static __device__ __forceinline__ v8f vzero8() {
  v8f z;
#pragma unroll
  for (int i = 0; i < 8; ++i) z[i] = 0.0f;
  return z;
}

static __device__ __forceinline__ v16bf combine16(v8bf lo, v8bf hi) {
  v16bf r;
#pragma unroll
  for (int i = 0; i < 8; ++i) { r[i] = lo[i]; r[i + 8] = hi[i]; }
  return r;
}

// A operand, 16x32 (MxK) bf16. lane<16: row M=lane, K = 0..15; lane>=16:
// K = 16..31 (K = 2*kch..2*kch+15, kch = (lane>=16)*8). `row` points at K=0.
static __device__ __forceinline__ v16bf load_a(const __bf16* row, int kch) {
  v8bf lo = *(const v8bf*)(row + 2 * kch);
  v8bf hi = *(const v8bf*)(row + 2 * kch + 8);
  return combine16(lo, hi);
}

// B operand, 32x16 (KxN) bf16. lane<16: column N=lane, K=0..15; lane>=16:
// column N=lane-16, K=16..31. `col` is pre-offset by (lane>=16)*16 in K.
static __device__ __forceinline__ v16bf load_bm(const __bf16* col) {
  v8bf lo = *(const v8bf*)(col);
  v8bf hi = *(const v8bf*)(col + 8);
  return combine16(lo, hi);
}

static __device__ __forceinline__ v8f wmma_bf16(v16bf a, v16bf b, v8f c) {
  return __builtin_amdgcn_wmma_f32_16x16x32_bf16(false, a, false, b,
                                                 (short)0, c, false, false);
}

// ------------------------------------------------------------------- TDM ----
// Issue TENSOR_LOAD_TO_LDS for a <=2D tile. D# per CDNA5 ISA ch.8:
//  group0: [1:0]=count=1, [63:32]=lds_addr, [120:64]=global_addr, [127:126]=2
//  group1: [17:16]=data_size(1=2B), [20]=pad_en, [24:22]=pad_interval,
//          [31:25]=pad_amount, [79:48]=tensor_dim0, [111:80]=tensor_dim1,
//          [127:112]=tile_dim0, [143:128]=tile_dim1, [207:160]=dim0_stride
//  groups 2/3: zero (tensor rank <= 2).
static __device__ __forceinline__ void tdm_load_2d(
    const void* gptr, unsigned lds_off,
    unsigned tensor_d0, unsigned tensor_d1,
    unsigned tile_d0, unsigned tile_d1,
    unsigned d0_stride,
    unsigned pad_interval, unsigned pad_amount) { // pad: ISA encodings
  unsigned long long ga = (unsigned long long)(size_t)gptr;
  u32x4 g0;
  g0[0] = 1u;                                  // count=1, user descriptor
  g0[1] = lds_off;                             // lds_addr (bytes)
  g0[2] = (unsigned)ga;                        // global_addr[31:0]
  g0[3] = (unsigned)((ga >> 32) & 0x01FFFFFFu) | (2u << 30); // [56:32] | type=2
  i32x8 g1;
  g1[0] = (int)((1u << 16) |                   // data_size = 2 bytes
                (1u << 20) |                   // pad_enable
                ((pad_interval & 7u) << 22) |
                ((pad_amount & 127u) << 25));
  g1[1] = (int)((tensor_d0 & 0xFFFFu) << 16);
  g1[2] = (int)(((tensor_d0 >> 16) & 0xFFFFu) | ((tensor_d1 & 0xFFFFu) << 16));
  g1[3] = (int)(((tensor_d1 >> 16) & 0xFFFFu) | ((tile_d0 & 0xFFFFu) << 16));
  g1[4] = (int)(tile_d1 & 0xFFFFu);            // tile_dim1 (0 = unused), dim2=0
  g1[5] = (int)d0_stride;                      // tensor_dim0_stride[31:0]
  g1[6] = 0;                                   // stride0[47:32], stride1[15:0]
  g1[7] = 0;
  i32x4 gz; gz[0] = 0; gz[1] = 0; gz[2] = 0; gz[3] = 0;
#if defined(__clang_major__) && (__clang_major__ >= 23)
  i32x8 gz8;
#pragma unroll
  for (int i = 0; i < 8; ++i) gz8[i] = 0;
  __builtin_amdgcn_tensor_load_to_lds(g0, g1, gz, gz, gz8, 0);
#else
  __builtin_amdgcn_tensor_load_to_lds(g0, g1, gz, gz, 0);
#endif
}

// ---------------------------------------------------------------- convert ---
__global__ __launch_bounds__(256) void f32_to_bf16_kernel(
    const float* __restrict__ src, __bf16* __restrict__ dst, int n) {
  int i = blockIdx.x * blockDim.x + threadIdx.x;
  if (i < n) dst[i] = (__bf16)src[i];
}

// ------------------------------------------------------------------- GEMM ---
// Y = X @ W^T.  X: [4096, 512] bf16.  W: [512, 512] bf16 (N x K row-major).
// MODE 0/1: bf16 out as [b,h,s,dk];  MODE 2: bf16 out as V^T [b,h,dk,s];
// MODE 3: f32 out to d_out [b,s,d].
template <int MODE>
__global__ __launch_bounds__(256) void gemm_bf16_kernel(
    const __bf16* __restrict__ X, const __bf16* __restrict__ W,
    __bf16* __restrict__ Ybf, float* __restrict__ Yf) {
  const int lane = threadIdx.x & 31;
  const int wave = threadIdx.x >> 5;
  const int hl   = lane >> 4;
  const int l15  = lane & 15;
  const int kch  = hl * 8;

  const int tileN = blockIdx.x & 7;
  const int tileM = blockIdx.x >> 3;
  const int m0 = tileM * 128 + wave * 16;
  const int n0 = tileN * 64;

  v8f acc[4];
#pragma unroll
  for (int j = 0; j < 4; ++j) acc[j] = vzero8();

  const __bf16* xrow = X + (size_t)(m0 + l15) * DDm;
  for (int k0 = 0; k0 < DDm; k0 += 32) {
    v16bf a = load_a(xrow + k0, kch);
#pragma unroll
    for (int j = 0; j < 4; ++j) {
      const __bf16* wcol = W + (size_t)(n0 + j * 16 + l15) * DDm + k0 + hl * 16;
      v16bf b = load_bm(wcol);
      acc[j] = wmma_bf16(a, b, acc[j]);
    }
  }

#pragma unroll
  for (int j = 0; j < 4; ++j) {
#pragma unroll
    for (int r = 0; r < 8; ++r) {
      const int m = m0 + r + hl * 8;
      const int n = n0 + j * 16 + l15;
      const float v = acc[j][r];
      if (MODE == 3) {
        Yf[(size_t)m * DDm + n] = v;
      } else {
        const int b  = m >> 11, s  = m & (SS - 1);
        const int h  = n >> 6,  dk = n & (DKK - 1);
        size_t idx;
        if (MODE == 2) idx = ((size_t)(b * HHn + h) * DKK + dk) * SS + s;
        else           idx = ((size_t)(b * HHn + h) * SS + s) * DKK + dk;
        Ybf[idx] = (__bf16)v;
      }
    }
  }
}

// -------------------------------------------------------- flash attention ---
// Grid: B*H*(S/128) = 256 blocks, 8 waves. Each wave owns a 16-row Q tile.
// Per 32-key step: wave 0 TDM-prefetches the next K (32x64) and V^T (64x32)
// tiles into the other LDS buffer, everyone computes 4 WMMAs for Q@K^T and
// 4 WMMAs for P@V from the current buffer, then s_wait_tensorcnt + barrier.
__global__ __launch_bounds__(256) void attn_kernel(
    const __bf16* __restrict__ Q,   // [b,h,s,dk]
    const __bf16* __restrict__ K,   // [b,h,s,dk]
    const __bf16* __restrict__ Vt,  // [b,h,dk,s]
    const float*  __restrict__ AW,  // [b,h,s,s]
    const unsigned char* __restrict__ mask, // [b,s] bool
    __bf16* __restrict__ Xout) {    // [b,s,d]
  __shared__ __bf16 kbuf[2][32 * KLD];     // 2 x 4.5 KB (padded rows)
  __shared__ __bf16 vbuf[2][64 * VLD];     // 2 x 5.0 KB (padded rows)
  __shared__ __bf16 lds_p[8 * 16 * 32];    // 8 KB: per-wave P staging tile

  const int lane = threadIdx.x & 31;
  const int wave = threadIdx.x >> 5;
  const int hl   = lane >> 4;
  const int l15  = lane & 15;
  const int kch  = hl * 8;

  const int bh   = blockIdx.x >> 4;  // b*H + h
  const int qblk = blockIdx.x & 15;
  const int b = bh >> 3, h = bh & 7;
  const int q0 = qblk * 128 + wave * 16;

  const __bf16* Qb  = Q  + (size_t)bh * SS * DKK;
  const __bf16* Kb  = K  + (size_t)bh * SS * DKK;
  const __bf16* Vb  = Vt + (size_t)bh * DKK * SS;
  const float*  AWb = AW + (size_t)bh * SS * SS;
  const unsigned char* mb = mask + (size_t)b * SS;

  const unsigned koff[2] = { (unsigned)(size_t)&kbuf[0][0],
                             (unsigned)(size_t)&kbuf[1][0] };
  const unsigned voff[2] = { (unsigned)(size_t)&vbuf[0][0],
                             (unsigned)(size_t)&vbuf[1][0] };

  // Q tile 16x64 -> two resident A operands.
  const v16bf aq0 = load_a(Qb + (size_t)(q0 + l15) * DKK,      kch);
  const v16bf aq1 = load_a(Qb + (size_t)(q0 + l15) * DKK + 32, kch);

  float mrow[8], lrow[8];
#pragma unroll
  for (int r = 0; r < 8; ++r) { mrow[r] = -__builtin_inff(); lrow[r] = 0.0f; }
  v8f accO[4];
#pragma unroll
  for (int j = 0; j < 4; ++j) accO[j] = vzero8();

  __bf16* pl = lds_p + wave * (16 * 32);

  // Prologue: DMA tile 0 into buffer 0.
  if (wave == 0) {
    // K: 1D stream of 32*64 elems; pad 16B after every 128B row (interval
    // code 4 = 32 dwords, amount code 3 = 4 dwords).
    tdm_load_2d(Kb, koff[0], 32 * DKK, 0, 32 * DKK, 0, 32 * DKK, 4u, 3u);
    // V^T: 2D tile 64 rows x 32 keys, row stride S; pad 16B after every 64B
    // row (interval code 3 = 16 dwords).
    tdm_load_2d(Vb, voff[0], 32, DKK, 32, DKK, SS, 3u, 3u);
    __builtin_amdgcn_s_wait_tensorcnt(0);
  }
  __syncthreads();

  for (int j0 = 0; j0 < SS; j0 += 32) {
    const int p = (j0 >> 5) & 1;
    if (wave == 0 && (j0 + 32) < SS) {
      tdm_load_2d(Kb + (size_t)(j0 + 32) * DKK, koff[1 - p],
                  32 * DKK, 0, 32 * DKK, 0, 32 * DKK, 4u, 3u);
      tdm_load_2d(Vb + (j0 + 32), voff[1 - p], 32, DKK, 32, DKK, SS, 3u, 3u);
    }
    const __bf16* kb = kbuf[p];
    const __bf16* vb = vbuf[p];

    // ---- scores: two 16x16 tiles over keys [j0, j0+32), operands from LDS
    v8f s0 = vzero8(), s1 = vzero8();
    {
      v16bf bk;
      bk = load_bm(kb + l15 * KLD + hl * 16);
      s0 = wmma_bf16(aq0, bk, s0);
      bk = load_bm(kb + l15 * KLD + 32 + hl * 16);
      s0 = wmma_bf16(aq1, bk, s0);
      bk = load_bm(kb + (l15 + 16) * KLD + hl * 16);
      s1 = wmma_bf16(aq0, bk, s1);
      bk = load_bm(kb + (l15 + 16) * KLD + 32 + hl * 16);
      s1 = wmma_bf16(aq1, bk, s1);
    }

    const int key0 = j0 + l15;
    const int key1 = j0 + 16 + l15;
    const bool mk0 = mb[key0] != 0;
    const bool mk1 = mb[key1] != 0;

    // Prefetch next attn_weight tile (global_prefetch_b8).
    if (j0 + 32 < SS)
      __builtin_prefetch(AWb + (size_t)(q0 + hl * 8) * SS + j0 + 32 + l15, 0, 1);

    // ---- scale, mask, bias
    float p0[8], p1[8];
#pragma unroll
    for (int r = 0; r < 8; ++r) {
      const int qi = q0 + r + hl * 8;
      const float* awr = AWb + (size_t)qi * SS;
      float v0 = s0[r] * 0.125f - awr[key0];
      float v1 = s1[r] * 0.125f - awr[key1];
      if (mk0) v0 = -__builtin_inff();
      if (mk1) v1 = -__builtin_inff();
      p0[r] = v0; p1[r] = v1;
    }

    // ---- online softmax; rows live across 16-lane halves of the C layout
#pragma unroll
    for (int r = 0; r < 8; ++r) {
      float rm = fmaxf(p0[r], p1[r]);
#pragma unroll
      for (int off = 1; off < 16; off <<= 1)
        rm = fmaxf(rm, __shfl_xor(rm, off, 32));
      const float mn = fmaxf(mrow[r], rm);
      float alpha = __expf(fminf(mrow[r] - mn, 0.0f));
      float e0 = __expf(p0[r] - mn);
      float e1 = __expf(p1[r] - mn);
      if (mn < -1e37f) { e0 = 0.0f; e1 = 0.0f; alpha = 1.0f; } // fully masked
      float rs = e0 + e1;
#pragma unroll
      for (int off = 1; off < 16; off <<= 1)
        rs += __shfl_xor(rs, off, 32);
      lrow[r] = lrow[r] * alpha + rs;
      mrow[r] = mn;
#pragma unroll
      for (int j = 0; j < 4; ++j) accO[j][r] = accO[j][r] * alpha;
      const int m = r + hl * 8;
      pl[m * 32 + l15]      = (__bf16)e0;
      pl[m * 32 + 16 + l15] = (__bf16)e1;
    }

    // P: C layout -> A operand via wave-private LDS (DS ops in-order per wave)
    const v16bf ap = load_a(pl + l15 * 32, kch);

    // ---- P @ V from the padded LDS V^T tile
#pragma unroll
    for (int j = 0; j < 4; ++j) {
      v16bf bv = load_bm(vb + (j * 16 + l15) * VLD + hl * 16);
      accO[j] = wmma_bf16(ap, bv, accO[j]);
    }

    if (wave == 0) __builtin_amdgcn_s_wait_tensorcnt(0);
    __syncthreads();
  }

  // ---- normalize and store bf16 to [b,s,d] for the Wo projection
#pragma unroll
  for (int r = 0; r < 8; ++r) {
    const float inv = lrow[r] > 0.0f ? 1.0f / lrow[r] : 0.0f;
    const int qi = q0 + r + hl * 8;
    __bf16* orow = Xout + ((size_t)b * SS + qi) * DDm + h * DKK;
#pragma unroll
    for (int j = 0; j < 4; ++j)
      orow[j * 16 + l15] = (__bf16)(accO[j][r] * inv);
  }
}

// ----------------------------------------------------------------- launch ---
extern "C" void kernel_launch(void* const* d_in, const int* in_sizes, int n_in,
                              void* d_out, int out_size, void* d_ws, size_t ws_size,
                              hipStream_t stream) {
  const float* query = (const float*)d_in[0];
  const float* key   = (const float*)d_in[1];
  const float* value = (const float*)d_in[2];
  const float* aw    = (const float*)d_in[3];
  const unsigned char* mask = (const unsigned char*)d_in[4]; // bool
  const float* Wq = (const float*)d_in[5];
  const float* Wk = (const float*)d_in[6];
  const float* Wv = (const float*)d_in[7];
  const float* Wo = (const float*)d_in[8];

  const size_t nX = (size_t)MTOT * DDm;   // 2,097,152 elems
  const size_t nW = (size_t)DDm * DDm;    //   262,144 elems

  // Workspace carve-out (~30 MB of bf16 buffers, all 16B-aligned).
  char* p = (char*)d_ws;
  __bf16* xq  = (__bf16*)p; p += nX * 2;
  __bf16* xk  = (__bf16*)p; p += nX * 2;
  __bf16* xv  = (__bf16*)p; p += nX * 2;
  __bf16* wqb = (__bf16*)p; p += nW * 2;
  __bf16* wkb = (__bf16*)p; p += nW * 2;
  __bf16* wvb = (__bf16*)p; p += nW * 2;
  __bf16* wob = (__bf16*)p; p += nW * 2;
  __bf16* qh  = (__bf16*)p; p += nX * 2;  // [b,h,s,dk]
  __bf16* kh  = (__bf16*)p; p += nX * 2;  // [b,h,s,dk]
  __bf16* vth = (__bf16*)p; p += nX * 2;  // [b,h,dk,s]
  __bf16* xo  = (__bf16*)p; p += nX * 2;  // attention output [b,s,d]

  const int cb = 256;
  f32_to_bf16_kernel<<<(int)((nX + cb - 1) / cb), cb, 0, stream>>>(query, xq, (int)nX);
  f32_to_bf16_kernel<<<(int)((nX + cb - 1) / cb), cb, 0, stream>>>(key,   xk, (int)nX);
  f32_to_bf16_kernel<<<(int)((nX + cb - 1) / cb), cb, 0, stream>>>(value, xv, (int)nX);
  f32_to_bf16_kernel<<<(int)((nW + cb - 1) / cb), cb, 0, stream>>>(Wq, wqb, (int)nW);
  f32_to_bf16_kernel<<<(int)((nW + cb - 1) / cb), cb, 0, stream>>>(Wk, wkb, (int)nW);
  f32_to_bf16_kernel<<<(int)((nW + cb - 1) / cb), cb, 0, stream>>>(Wv, wvb, (int)nW);
  f32_to_bf16_kernel<<<(int)((nW + cb - 1) / cb), cb, 0, stream>>>(Wo, wob, (int)nW);

  gemm_bf16_kernel<0><<<256, 256, 0, stream>>>(xq, wqb, qh,  nullptr);
  gemm_bf16_kernel<1><<<256, 256, 0, stream>>>(xk, wkb, kh,  nullptr);
  gemm_bf16_kernel<2><<<256, 256, 0, stream>>>(xv, wvb, vth, nullptr);

  attn_kernel<<<256, 256, 0, stream>>>(qh, kh, vth, aw, mask, xo);

  gemm_bf16_kernel<3><<<256, 256, 0, stream>>>(xo, wob, nullptr, (float*)d_out);
}
